// HebbianPretrainedHopfield_25151328485834
// MI455X (gfx1250) — matrix operand order repaired and checked by reading, not compiled, not run on hardware
//
#include <hip/hip_runtime.h>

// Problem constants (from reference)
#define NB 4
#define NL 4096
#define ND 1024
#define NH 16
#define NM 512
#define NHD 64
#define NROWS (NB * NL)          // 16384 rows of D for the projections
#define QROWS (NB * NL * NH)     // 262144 rows of HD for the Hopfield stage
#define BETA 16.0f

typedef __attribute__((ext_vector_type(16))) __bf16 v16bf;
typedef __attribute__((ext_vector_type(8)))  __bf16 v8bf;
typedef __attribute__((ext_vector_type(4)))  __bf16 v4bf;
typedef __attribute__((ext_vector_type(8)))  float  v8f;
typedef __attribute__((ext_vector_type(4)))  unsigned int v4u;
typedef __attribute__((ext_vector_type(8)))  int v8i;
typedef __attribute__((ext_vector_type(4)))  int v4i;

static __device__ __forceinline__ v8f wmma_bf(v16bf a, v16bf b, v8f c) {
  // D = A(16x32 bf16) * B(32x16 bf16) + C(16x16 f32)
  return __builtin_amdgcn_wmma_f32_16x16x32_bf16(false, a, false, b, (short)0, c, false, false);
}

// A-operand fragment (16x32 bf16) from a row-major [rows x ld] bf16 matrix.
// lane<16: row m, K = {k0..k0+7, k0+16..k0+23}; lane>=16: K = {+8..+15, +24..+31}.
static __device__ __forceinline__ v16bf load_fragA_g(const __bf16* base, int ld,
                                                     int m0, int k0, int lane) {
  int r = m0 + (lane & 15);
  int k = k0 + ((lane >> 4) << 3);
  const __bf16* p = base + (size_t)r * ld + k;
  v8bf lo = *(const v8bf*)p;
  v8bf hi = *(const v8bf*)(p + 16);
  return __builtin_shufflevector(lo, hi, 0,1,2,3,4,5,6,7,8,9,10,11,12,13,14,15);
}

// B-operand fragment (32x16 bf16). Source is B^T stored row-major (row n = col n of B).
// Per-lane: column n = n0 + (lane&15), K = {k0..k0+7, k0+16..k0+23}; lane>=16: K = {+8..+15, +24..+31}.
static __device__ __forceinline__ v16bf load_fragB_g(const __bf16* baseT, int ld,
                                                     int n0, int k0, int lane) {
  int n = n0 + (lane & 15);
  int k = k0 + ((lane >> 4) << 3);
  const __bf16* p = baseT + (size_t)n * ld + k;
  v8bf lo = *(const v8bf*)p;
  v8bf hi = *(const v8bf*)(p + 16);
  return __builtin_shufflevector(lo, hi, 0,1,2,3,4,5,6,7,8,9,10,11,12,13,14,15);
}

// ---------------------------------------------------------------- converts
__global__ void cvt_f32_bf16(const float* __restrict__ src,
                             __bf16* __restrict__ dst, int n) {
  int i = (blockIdx.x * blockDim.x + threadIdx.x) * 4;
  if (i + 3 < n) {
    float4 v = *(const float4*)(src + i);
    v4bf o = { (__bf16)v.x, (__bf16)v.y, (__bf16)v.z, (__bf16)v.w };
    *(v4bf*)(dst + i) = o;
  }
}

__global__ void prep_mem(const float* __restrict__ mem,
                         __bf16* __restrict__ memb,   // (M x HD) bf16
                         __bf16* __restrict__ memt) { // (HD x M) bf16 (= mem^T)
  int idx = blockIdx.x * blockDim.x + threadIdx.x;
  if (idx < NM * NHD) {
    float v = mem[idx];
    memb[idx] = (__bf16)v;
    int m = idx >> 6, d = idx & 63;
    memt[d * NM + m] = (__bf16)v;
  }
}

// ---------------------------------------------------------------- GEMM
// out(n_rows x N) = A(n_rows x K, bf16) @ W^T + bias ; W is (N x K) row-major.
// One wave computes a 64(M) x 64(N) tile: 16 WMMA per 16x16B loads per k-step.
template <int OUT_BF16>
__global__ void gemm_bf16(const __bf16* __restrict__ A,
                          const __bf16* __restrict__ W,
                          const float* __restrict__ bias,
                          void* __restrict__ out,
                          int n_rows, int N, int K) {
  int wid  = (blockIdx.x * blockDim.x + threadIdx.x) >> 5;
  int lane = threadIdx.x & 31;
  int n_nt = N >> 6;
  int mt = wid / n_nt;
  int nt = wid - mt * n_nt;
  int m0 = mt * 64, n0 = nt * 64;
  if (m0 >= n_rows) return;  // wave-uniform

  v8f c[4][4] = {};
  for (int k0 = 0; k0 < K; k0 += 32) {
    v16bf a[4];
#pragma unroll
    for (int i = 0; i < 4; ++i)
      a[i] = load_fragA_g(A, K, m0 + i * 16, k0, lane);
    if (k0 + 128 < K) {
      // prefetch the streaming A operand four k-tiles ahead (global_prefetch_b8)
      const __bf16* pf = A + (size_t)(m0 + (lane & 15)) * K + (k0 + 128);
      __builtin_prefetch(pf, 0, 3);
    }
#pragma unroll
    for (int j = 0; j < 4; ++j) {
      v16bf b = load_fragB_g(W, K, n0 + j * 16, k0, lane);
      c[0][j] = wmma_bf(a[0], b, c[0][j]);
      c[1][j] = wmma_bf(a[1], b, c[1][j]);
      c[2][j] = wmma_bf(a[2], b, c[2][j]);
      c[3][j] = wmma_bf(a[3], b, c[3][j]);
    }
  }
  int colw = lane & 15;
  int rb   = (lane >> 4) << 3;
  for (int j = 0; j < 4; ++j) {
    int col = n0 + j * 16 + colw;
    float bv = bias[col];
    for (int i = 0; i < 4; ++i) {
      for (int vg = 0; vg < 8; ++vg) {
        int row = m0 + i * 16 + rb + vg;
        float v = c[i][j][vg] + bv;
        if (OUT_BF16) ((__bf16*)out)[(size_t)row * N + col] = (__bf16)v;
        else          ((float*)out)[(size_t)row * N + col] = v;
      }
    }
  }
}

// ---------------------------------------------------------------- Hopfield
// One wave owns a 16-row tile of q (16 x 64). Two fused iterations:
//   E(16x512) = q @ mem^T  (WMMA, beta-scaled into LDS f32)
//   softmax over 512 per row, then q'(16x64) = softmax @ mem (WMMA).
// mem (M x HD) and mem^T (HD x M), 128 KB total, are DMA'd once per block
// into LDS by the Tensor Data Mover; all B-operand reads then hit LDS.
// Per-wave LDS: e[16][512] f32 (32KB) + w[16][512] bf16 (16KB) + qs[16][64] f32 (4KB)
#define HOP_WAVE_LDS (32768 + 16384 + 4096)   // 53248
#define HOP_WAVES 2
#define HOP_MEM_OFF (HOP_WAVES * HOP_WAVE_LDS) // 106496
#define HOP_LDS_TOTAL (HOP_MEM_OFF + 131072)   // 237568 (< 320KB/WG)

__global__ void hopfield_kernel(__bf16* __restrict__ q,
                                const __bf16* __restrict__ memb,   // memb||memt contiguous
                                const __bf16* __restrict__ memt) {
  extern __shared__ char smem[];
  int wave = threadIdx.x >> 5;
  int lane = threadIdx.x & 31;
  char* wbase = smem + (size_t)wave * HOP_WAVE_LDS;
  float*  e  = (float*)wbase;
  __bf16* wb = (__bf16*)(wbase + 32768);
  float*  qs = (float*)(wbase + 32768 + 16384);
  const __bf16* lmb = (const __bf16*)(smem + HOP_MEM_OFF);   // mem   (512 x 64)
  const __bf16* lmt = lmb + (NM * NHD);                      // mem^T (64 x 512)

  // --- TDM: one 128KB 1-D tile (16384 x 8B) global -> LDS, issued by wave 0
  if (wave == 0) {
    unsigned long long ga = (unsigned long long)memb;
    v4u g0;
    g0[0] = 1u;                                  // count=1, user descriptor
    g0[1] = (unsigned)HOP_MEM_OFF;               // lds_addr (bytes)
    g0[2] = (unsigned)(ga & 0xffffffffu);        // global_addr[31:0]
    g0[3] = (unsigned)((ga >> 32) & 0x1ffffffu)  // global_addr[56:32]
          | (2u << 30);                          // type = 2 ("image")
    const unsigned elems = 131072u / 8u;         // 16384 x 8-byte elements
    v8i g1;
    g1[0] = (int)(3u << 16);                     // wg_mask=0, data_size=3 (8B)
    g1[1] = (int)((elems & 0xffffu) << 16);      // tensor_dim0[15:0]
    g1[2] = (int)((elems >> 16) | (1u << 16));   // tensor_dim0[31:16] | tensor_dim1=1
    g1[3] = (int)((elems & 0xffffu) << 16);      // tile_dim0 = 16384
    g1[4] = (int)1;                              // tile_dim1=1, tile_dim2=0
    g1[5] = (int)elems;                          // tensor_dim0_stride[31:0]
    g1[6] = 0;                                   // stride0[47:32] | stride1[15:0]
    g1[7] = 0;                                   // stride1[47:16]
    v4i gz4 = {};
    v8i gz8 = {};
    __builtin_amdgcn_tensor_load_to_lds(g0, g1, gz4, gz4, gz8, 0);
  }

  int tile = blockIdx.x * HOP_WAVES + wave;  // QROWS/16 tiles, exact grid
  size_t row0 = (size_t)tile * 16;

  // stage q tile -> qs (f32) while the TDM streams mem into LDS
  for (int i = lane; i < 16 * 64; i += 32)
    qs[i] = (float)q[row0 * 64 + i];
  if (wave == 0) __builtin_amdgcn_s_wait_tensorcnt(0);
  __syncthreads();

  int ar = lane & 15;             // A row within tile
  int ak = (lane >> 4) << 3;      // A k sub-offset (0 or 8)

  for (int it = 0; it < 2; ++it) {
    // A frags for energy (q tile, K=64 -> two 16x32 frags), converted f32->bf16
    v16bf a0, a1;
    for (int j = 0; j < 8; ++j) {
      a0[j]     = (__bf16)qs[ar * 64 + ak + j];
      a0[j + 8] = (__bf16)qs[ar * 64 + 16 + ak + j];
      a1[j]     = (__bf16)qs[ar * 64 + 32 + ak + j];
      a1[j + 8] = (__bf16)qs[ar * 64 + 48 + ak + j];
    }
    // E = beta * (q @ mem^T): 32 n-tiles of 16 over M=512; B frags from LDS
    for (int mt = 0; mt < 32; ++mt) {
      v16bf b0 = load_fragB_g(lmb, NHD, mt * 16, 0,  lane);
      v16bf b1 = load_fragB_g(lmb, NHD, mt * 16, 32, lane);
      v8f c = {};
      c = wmma_bf(a0, b0, c);
      c = wmma_bf(a1, b1, c);
      int col = mt * 16 + (lane & 15);
      int rb  = (lane >> 4) << 3;
      for (int vg = 0; vg < 8; ++vg)
        e[(rb + vg) * NM + col] = BETA * c[vg];
    }
    __syncthreads();

    // softmax per row (512 wide): lane r -> cols [0,256), lane r+16 -> [256,512)
    {
      int r = lane & 15, h = lane >> 4;
      float* er = e + r * NM + h * 256;
      float mx = -3.0e38f;
      for (int j = 0; j < 256; ++j) mx = fmaxf(mx, er[j]);
      mx = fmaxf(mx, __shfl_xor(mx, 16, 32));
      float sum = 0.0f;
      for (int j = 0; j < 256; ++j) {
        float v = __expf(er[j] - mx);
        er[j] = v;
        sum += v;
      }
      sum += __shfl_xor(sum, 16, 32);
      float rs = 1.0f / sum;
      __bf16* wr = wb + r * NM + h * 256;
      for (int j = 0; j < 256; ++j) wr[j] = (__bf16)(er[j] * rs);
    }
    __syncthreads();

    // q' = w(16x512) @ mem(512x64); B^T = mem^T (64 x 512) from LDS
    v8f cr[4] = {};
    for (int kt = 0; kt < 16; ++kt) {
      int koff = kt * 32 + ak;
      const __bf16* pa = wb + ar * NM + koff;
      v8bf lo = *(const v8bf*)pa;
      v8bf hi = *(const v8bf*)(pa + 16);
      v16bf a = __builtin_shufflevector(lo, hi, 0,1,2,3,4,5,6,7,8,9,10,11,12,13,14,15);
      for (int ntl = 0; ntl < 4; ++ntl) {
        v16bf b = load_fragB_g(lmt, NM, ntl * 16, kt * 32, lane);
        cr[ntl] = wmma_bf(a, b, cr[ntl]);
      }
    }
    __syncthreads();
    // scatter C -> qs
    for (int ntl = 0; ntl < 4; ++ntl) {
      int col = ntl * 16 + (lane & 15);
      int rb  = (lane >> 4) << 3;
      for (int vg = 0; vg < 8; ++vg)
        qs[(rb + vg) * 64 + col] = cr[ntl][vg];
    }
    __syncthreads();
  }

  // write updated tile back (bf16, feeds output projection GEMM)
  for (int i = lane; i < 16 * 64; i += 32)
    q[row0 * 64 + i] = (__bf16)qs[i];
}

// ---------------------------------------------------------------- launcher
extern "C" void kernel_launch(void* const* d_in, const int* in_sizes, int n_in,
                              void* d_out, int out_size, void* d_ws, size_t ws_size,
                              hipStream_t stream) {
  (void)in_sizes; (void)n_in; (void)out_size; (void)ws_size;
  const float* x   = (const float*)d_in[0];
  const float* Wq  = (const float*)d_in[1];
  const float* bq  = (const float*)d_in[2];
  const float* Wo  = (const float*)d_in[3];
  const float* bo  = (const float*)d_in[4];
  const float* mem = (const float*)d_in[5];
  float* out = (float*)d_out;

  // workspace layout (bf16 staging), ~71.4 MB total; memb||memt contiguous for TDM
  char* ws = (char*)d_ws;
  const size_t XB  = (size_t)NROWS * ND * 2;      // 33554432
  const size_t WB  = (size_t)ND * ND * 2;         // 2097152
  const size_t MB  = (size_t)NM * NHD * 2;        // 65536
  __bf16* xb   = (__bf16*)(ws);
  __bf16* wqb  = (__bf16*)(ws + XB);
  __bf16* wob  = (__bf16*)(ws + XB + WB);
  __bf16* qb   = (__bf16*)(ws + XB + 2 * WB);
  __bf16* memb = (__bf16*)(ws + 2 * XB + 2 * WB);
  __bf16* memt = (__bf16*)(ws + 2 * XB + 2 * WB + MB);

  // 1) fp32 -> bf16 staging
  cvt_f32_bf16<<<16384, 256, 0, stream>>>(x,  xb,  NROWS * ND);
  cvt_f32_bf16<<<1024,  256, 0, stream>>>(Wq, wqb, ND * ND);
  cvt_f32_bf16<<<1024,  256, 0, stream>>>(Wo, wob, ND * ND);
  prep_mem<<<128, 256, 0, stream>>>(mem, memb, memt);

  // 2) q = x @ Wq^T + bq   (bf16 out); 64x64 wave tiles: 256*16 = 4096 waves
  gemm_bf16<1><<<512, 256, 0, stream>>>(xb, wqb, bq, (void*)qb, NROWS, ND, ND);

  // 3) two fused Hopfield iterations, in-place on qb
  //    QROWS/16 = 16384 tiles, 2 waves/block -> 8192 blocks
  hopfield_kernel<<<8192, 32 * HOP_WAVES, HOP_LDS_TOTAL, stream>>>(qb, memb, memt);

  // 4) out = q @ Wo^T + bo  (fp32 out)
  gemm_bf16<0><<<512, 256, 0, stream>>>(qb, wob, bo, (void*)out, NROWS, ND, ND);
}
